// SimpleMoE_2860448219600
// MI455X (gfx1250) — matrix operand order repaired and checked by reading, not compiled, not run on hardware
//
#include <hip/hip_runtime.h>
#include <hip/hip_bf16.h>

typedef __bf16 bf16;
typedef __attribute__((ext_vector_type(16))) __bf16 v16bf;
typedef __attribute__((ext_vector_type(8)))  float  v8f;

#define T_TOK 8192
#define HID   1024
#define FFD   4096
#define NEXP  8

#if __has_builtin(__builtin_amdgcn_global_load_async_to_lds_b128)
#define HAS_ASYNC_LDS 1
#endif

__device__ __forceinline__ unsigned short f2bfbits(float f) {
    unsigned u = __float_as_uint(f);
    return (unsigned short)((u + 0x7FFFu + ((u >> 16) & 1u)) >> 16);
}
__device__ __forceinline__ bf16 f2bf(float f) {
    unsigned short s = f2bfbits(f);
    return __builtin_bit_cast(bf16, s);
}
__device__ __forceinline__ v16bf pack16(uint4 a, uint4 b) {
    union { uint4 u[2]; v16bf v; } U; U.u[0] = a; U.u[1] = b; return U.v;
}

#ifdef HAS_ASYNC_LDS
typedef int v4i __attribute__((vector_size(16)));
typedef v4i __attribute__((address_space(1))) as1_v4i;
typedef v4i __attribute__((address_space(3))) as3_v4i;
__device__ __forceinline__ void async_copy16(const bf16* g, bf16* l) {
    __builtin_amdgcn_global_load_async_to_lds_b128((as1_v4i*)g, (as3_v4i*)l, 0, 0);
}
__device__ __forceinline__ void wait_async() {
#if __has_builtin(__builtin_amdgcn_s_wait_asynccnt)
    __builtin_amdgcn_s_wait_asynccnt(0);
#else
    asm volatile("s_wait_asynccnt 0x0" ::: "memory");
#endif
}
#endif

// ---------------------------------------------------------------- x -> bf16
__global__ __launch_bounds__(256) void cvt_kernel(const float* __restrict__ x,
                                                  bf16* __restrict__ xb) {
    size_t i = ((size_t)blockIdx.x * 256 + threadIdx.x) * 4;
    float4 f = *(const float4*)(x + i);
    ushort4 u;
    u.x = f2bfbits(f.x); u.y = f2bfbits(f.y);
    u.z = f2bfbits(f.z); u.w = f2bfbits(f.w);
    *(ushort4*)((unsigned short*)xb + i) = u;
}

// ---------------------------------------------------------------- gating
__global__ __launch_bounds__(256) void gate_kernel(
    const float* __restrict__ x, const float* __restrict__ gW,
    const float* __restrict__ gb, float* __restrict__ wcomb,
    int* __restrict__ tok_list, int* __restrict__ cnt_i,
    float* __restrict__ probs_part) {
    __shared__ float pw[8][NEXP];
    int tid = threadIdx.x, lane = tid & 31, wave = tid >> 5;
    int t = blockIdx.x * 8 + wave;
    float a[NEXP] = {0.f, 0.f, 0.f, 0.f, 0.f, 0.f, 0.f, 0.f};
    const float* xr = x + (size_t)t * HID;
    for (int h = lane; h < HID; h += 32) {
        float xv = xr[h];
#pragma unroll
        for (int e = 0; e < NEXP; e++) a[e] += xv * gW[e * HID + h];
    }
#pragma unroll
    for (int e = 0; e < NEXP; e++)
        for (int off = 16; off > 0; off >>= 1) a[e] += __shfl_xor(a[e], off);
    if (lane == 0) {
        float lg[NEXP];
#pragma unroll
        for (int e = 0; e < NEXP; e++) lg[e] = a[e] + gb[e];
        int i0 = 0;
        for (int e = 1; e < NEXP; e++) if (lg[e] > lg[i0]) i0 = e;
        int i1 = (i0 == 0) ? 1 : 0;
        for (int e = 0; e < NEXP; e++) if (e != i0 && lg[e] > lg[i1]) i1 = e;
        float m2 = fmaxf(lg[i0], lg[i1]);
        float e0 = expf(lg[i0] - m2), e1 = expf(lg[i1] - m2);
        float inv = 1.0f / (e0 + e1);
        float w0 = e0 * inv, w1 = e1 * inv;
#pragma unroll
        for (int e = 0; e < NEXP; e++)
            wcomb[(size_t)t * NEXP + e] = (e == i0) ? w0 : ((e == i1) ? w1 : 0.0f);
        int p0 = atomicAdd(&cnt_i[i0], 1); tok_list[i0 * T_TOK + p0] = t;
        int p1 = atomicAdd(&cnt_i[i1], 1); tok_list[i1 * T_TOK + p1] = t;
        float mx = lg[0];
        for (int e = 1; e < NEXP; e++) mx = fmaxf(mx, lg[e]);
        float s = 0.f, pe[NEXP];
#pragma unroll
        for (int e = 0; e < NEXP; e++) { pe[e] = expf(lg[e] - mx); s += pe[e]; }
        float is = 1.0f / s;
#pragma unroll
        for (int e = 0; e < NEXP; e++) pw[wave][e] = pe[e] * is;
    }
    __syncthreads();
    if (tid == 0) {
#pragma unroll
        for (int e = 0; e < NEXP; e++) {
            float s = 0.f;
            for (int w = 0; w < 8; w++) s += pw[w][e];   // fixed order: deterministic
            probs_part[(size_t)blockIdx.x * NEXP + e] = s;
        }
    }
}

// ---------------------------------------------------------------- finalize
__global__ void finalize_kernel(const int* __restrict__ cnt_i, int* __restrict__ offs,
                                const float* __restrict__ probs_part,
                                float* __restrict__ out_tail) {
    if (threadIdx.x == 0 && blockIdx.x == 0) {
        int o = 0;
        for (int e = 0; e < NEXP; e++) { offs[e] = o; o += cnt_i[e]; }
        offs[NEXP] = o;
        float l = 0.f;
        for (int e = 0; e < NEXP; e++) {
            float s = 0.f;
            for (int b = 0; b < T_TOK / 8; b++) s += probs_part[(size_t)b * NEXP + e];
            float p = s / (float)T_TOK;
            l += p * p;
        }
        out_tail[0] = l * (float)NEXP;
        for (int e = 0; e < NEXP; e++) out_tail[1 + e] = (float)cnt_i[e];
    }
}

// ---------------------------------------------------------------- WMMA GEMM
// 128x128 tile per 256-thread workgroup (8 waves, 2x4 grid, 64x32 per wave).
// Double-buffered LDS: async A gather-copy (ASYNCcnt) + fp32->bf16 B convert for
// tile k+1 overlap the 16 WMMAs of tile k; one barrier per K-iteration.
template <int KTOT, int NTOT, bool GELU_OUT>
__global__ __launch_bounds__(256) void moe_gemm(
    const bf16* __restrict__ A, const float* __restrict__ W,
    const float* __restrict__ bias, const int* __restrict__ tok_list,
    const int* __restrict__ cnt_i, const int* __restrict__ offs,
    const float* __restrict__ wcomb, bf16* __restrict__ Hact,
    float* __restrict__ Out) {
    __shared__ bf16 Alds[2][128 * 64];
    __shared__ bf16 Blds[2][128 * 64];
    __shared__ int rowsrc[128];
    __shared__ int toks[128];
    __shared__ float wrow[128];

    const int e = blockIdx.z, mtile = blockIdx.y, ntile = blockIdx.x;
    const int cnt = cnt_i[e];
    if (mtile * 128 >= cnt) return;
    const int off = offs[e];
    const int tid = threadIdx.x;

    if (tid < 128) {
        int lrow = mtile * 128 + tid;
        int cl = lrow < cnt ? lrow : cnt - 1;        // clamp padding rows to valid data
        int t = tok_list[e * T_TOK + cl];
        if (GELU_OUT) {
            rowsrc[tid] = t;                          // gather x rows by token
        } else {
            rowsrc[tid] = off + cl;                   // gather compact activation slots
            toks[tid] = t;
            wrow[tid] = wcomb[(size_t)t * NEXP + e];
        }
    }
    __syncthreads();

    const int lane = tid & 31, wave = tid >> 5;
    const int waveM = wave >> 2, waveN = wave & 3;
    const int m_off = waveM * 64, n_off = waveN * 32;
    const int lm = lane & 15, hi = lane >> 4;

    v8f acc[4][2] = {};

    const int rtA = tid >> 3, cbA = (tid & 7) * 8;
    const int rwB = tid >> 5, c4B = (tid & 31) * 4;
    const float* Wt = W + (size_t)e * KTOT * NTOT + (size_t)ntile * 128;

    auto load_tiles = [&](int k0, int buf) {
#pragma unroll
        for (int rb = 0; rb < 128; rb += 32) {       // A: 128x64 bf16, gathered rows
            int r = rb + rtA;
            const bf16* src = A + (size_t)rowsrc[r] * KTOT + k0 + cbA;
#ifdef HAS_ASYNC_LDS
            async_copy16(src, &Alds[buf][r * 64 + cbA]);
#else
            *(uint4*)&Alds[buf][r * 64 + cbA] = *(const uint4*)src;
#endif
        }
        if (k0 + 64 < KTOT)                           // hint next weight tile
            __builtin_prefetch(Wt + (size_t)(k0 + 64) * NTOT + c4B, 0, 1);
#pragma unroll
        for (int rb = 0; rb < 64; rb += 8) {         // B: 64x128 fp32 -> bf16, [n][k]
            int r = rb + rwB;
            float4 f = *(const float4*)(Wt + (size_t)(k0 + r) * NTOT + c4B);
            Blds[buf][(c4B + 0) * 64 + r] = f2bf(f.x);
            Blds[buf][(c4B + 1) * 64 + r] = f2bf(f.y);
            Blds[buf][(c4B + 2) * 64 + r] = f2bf(f.z);
            Blds[buf][(c4B + 3) * 64 + r] = f2bf(f.w);
        }
    };

    load_tiles(0, 0);
    int p = 0;
    for (int k0 = 0; k0 < KTOT; k0 += 64) {
#ifdef HAS_ASYNC_LDS
        wait_async();                                 // ASYNCcnt not covered by barrier
#endif
        __syncthreads();
        if (k0 + 64 < KTOT) load_tiles(k0 + 64, p ^ 1);

#pragma unroll
        for (int ks = 0; ks < 64; ks += 32) {
            v16bf af[4], bfr[2];
#pragma unroll
            for (int mt = 0; mt < 4; mt++) {
                // A layout: elems 0..7 -> k = ks + hi*8 + i ; elems 8..15 -> +16
                const bf16* pa = &Alds[p][(m_off + mt * 16 + lm) * 64 + ks + hi * 8];
                af[mt] = pack16(*(const uint4*)pa, *(const uint4*)(pa + 16));
            }
#pragma unroll
            for (int nt = 0; nt < 2; nt++) {
                // B layout: elems 0..7 -> k = ks + hi*8 + i ; elems 8..15 -> +16
                const bf16* pb = &Blds[p][(n_off + nt * 16 + lm) * 64 + ks + hi * 8];
                bfr[nt] = pack16(*(const uint4*)pb, *(const uint4*)(pb + 16));
            }
#pragma unroll
            for (int mt = 0; mt < 4; mt++)
#pragma unroll
                for (int nt = 0; nt < 2; nt++)
                    acc[mt][nt] = __builtin_amdgcn_wmma_f32_16x16x32_bf16(
                        false, af[mt], false, bfr[nt], (short)0, acc[mt][nt],
                        false, false);
        }
        p ^= 1;
    }

    const float* be = bias + (size_t)e * NTOT;
#pragma unroll
    for (int mt = 0; mt < 4; mt++) {
#pragma unroll
        for (int nt = 0; nt < 2; nt++) {
            v8f c = acc[mt][nt];
            int n_l = n_off + nt * 16 + lm;
            int ng = ntile * 128 + n_l;
            float bv = be[ng];
#pragma unroll
            for (int r = 0; r < 8; r++) {
                int m_l = m_off + mt * 16 + r + hi * 8;   // C layout: m = r + 8*hi
                int lrow = mtile * 128 + m_l;
                if (lrow < cnt) {
                    float v = c[r] + bv;
                    if (GELU_OUT) {
                        v = 0.5f * v * (1.0f + erff(v * 0.70710678118654752f));
                        Hact[(size_t)(off + lrow) * NTOT + ng] = f2bf(v);
                    } else {
                        atomicAdd(&Out[(size_t)toks[m_l] * HID + ng], wrow[m_l] * v);
                    }
                }
            }
        }
    }
}

// ---------------------------------------------------------------- launch
extern "C" void kernel_launch(void* const* d_in, const int* in_sizes, int n_in,
                              void* d_out, int out_size, void* d_ws, size_t ws_size,
                              hipStream_t stream) {
    (void)in_sizes; (void)n_in; (void)ws_size;
    const float* x  = (const float*)d_in[0];
    const float* gW = (const float*)d_in[1];
    const float* gb = (const float*)d_in[2];
    const float* W1 = (const float*)d_in[3];
    const float* b1 = (const float*)d_in[4];
    const float* W2 = (const float*)d_in[5];
    const float* b2 = (const float*)d_in[6];
    float* out = (float*)d_out;

    // workspace layout (all re-initialized every call)
    char* ws = (char*)d_ws;
    const size_t SZ_XBF   = (size_t)T_TOK * HID * 2;        // 16 MB
    const size_t SZ_HACT  = (size_t)2 * T_TOK * FFD * 2;    // 128 MB
    const size_t SZ_WCOMB = (size_t)T_TOK * NEXP * 4;       // 256 KB
    const size_t SZ_TLIST = (size_t)NEXP * T_TOK * 4;       // 256 KB
    const size_t SZ_CNT   = 256;
    const size_t SZ_OFFS  = 256;
    size_t o = 0;
    bf16*  Xbf       = (bf16*)(ws + o);  o += SZ_XBF;
    bf16*  Hact      = (bf16*)(ws + o);  o += SZ_HACT;
    float* wcomb     = (float*)(ws + o); o += SZ_WCOMB;
    int*   tok_list  = (int*)(ws + o);   o += SZ_TLIST;
    int*   cnt_i     = (int*)(ws + o);   o += SZ_CNT;
    int*   offs      = (int*)(ws + o);   o += SZ_OFFS;
    float* probs_prt = (float*)(ws + o);

    (void)hipMemsetAsync(out, 0, (size_t)out_size * sizeof(float), stream);
    (void)hipMemsetAsync(cnt_i, 0, SZ_CNT, stream);

    cvt_kernel<<<(T_TOK * HID) / (256 * 4), 256, 0, stream>>>(x, Xbf);
    gate_kernel<<<T_TOK / 8, 256, 0, stream>>>(x, gW, gb, wcomb, tok_list, cnt_i,
                                               probs_prt);
    finalize_kernel<<<1, 32, 0, stream>>>(cnt_i, offs, probs_prt,
                                          out + (size_t)T_TOK * HID);
    // FFN1: [slots x 1024] @ [1024 x 4096] -> GELU -> Hact (bf16)
    moe_gemm<HID, FFD, true><<<dim3(FFD / 128, T_TOK / 128, NEXP), 256, 0, stream>>>(
        Xbf, W1, b1, tok_list, cnt_i, offs, nullptr, Hact, nullptr);
    // FFN2: [slots x 4096] @ [4096 x 1024] -> scaled atomic add into out
    moe_gemm<FFD, HID, false><<<dim3(HID / 128, T_TOK / 128, NEXP), 256, 0, stream>>>(
        Hact, W2, b2, tok_list, cnt_i, offs, wcomb, nullptr, out);
}
